// Attention_16690242913193
// MI455X (gfx1250) — matrix operand order repaired and checked by reading, not compiled, not run on hardware
//
#include <hip/hip_runtime.h>

// ---------------------------------------------------------------------------
// Types for CDNA5 WMMA (wave32): v_wmma_f32_16x16x32_bf16
// ---------------------------------------------------------------------------
typedef __bf16 bf16_t;
typedef __bf16 v16bf __attribute__((ext_vector_type(16)));
typedef __bf16 bf8v  __attribute__((ext_vector_type(8)));
typedef float  v8f   __attribute__((ext_vector_type(8)));

#define SEQ   2048
#define DIM   1024
#define NHEAD 16
#define HD    64
#define BATCH 2
#define ROWS  (BATCH*SEQ)     // 4096

__device__ __forceinline__ v8f wmma_bf16(v16bf a, v16bf b, v8f c) {
  // D = A(16x32 bf16) x B(32x16 bf16) + C(16x16 f32)
  return __builtin_amdgcn_wmma_f32_16x16x32_bf16(
      false, a, false, b, (short)0, c, false, false);
}

__device__ __forceinline__ v16bf make_frag(const bf16_t* lo, const bf16_t* hi) {
  union { v16bf v; bf8v h[2]; } u;
  u.h[0] = *(const bf8v*)lo;
  u.h[1] = *(const bf8v*)hi;
  return u.v;
}
// A fragment (16x32, row-major): lane m=lane&15, h=lane>>4
//   elems 0..15 -> K=16h+i
__device__ __forceinline__ v16bf load_a_frag(const bf16_t* row_k0, int lh) {
  return make_frag(row_k0 + 16 * lh, row_k0 + 16 * lh + 8);
}
// B fragment (32x16) from N-major buffer: elems 0..15 -> K=16h+i
__device__ __forceinline__ v16bf load_b_frag(const bf16_t* nrow_k0, int lh) {
  return make_frag(nrow_k0 + 16 * lh, nrow_k0 + 16 * lh + 8);
}

// ---------------------------------------------------------------------------
// Kernel A: f32 -> bf16 convert (optionally transposed for N-major weights)
// ---------------------------------------------------------------------------
__global__ void cvt_kernel(const float* __restrict__ src, bf16_t* __restrict__ dst, int n) {
  int i = blockIdx.x * blockDim.x + threadIdx.x;
  if (i < n) dst[i] = (bf16_t)src[i];
}
__global__ void cvt_t_kernel(const float* __restrict__ src, bf16_t* __restrict__ dst,
                             int rows, int cols) {
  int i = blockIdx.x * blockDim.x + threadIdx.x;
  if (i < rows * cols) {
    int r = i / cols, c = i % cols;
    dst[(size_t)c * rows + r] = (bf16_t)src[i];
  }
}

// ---------------------------------------------------------------------------
// Kernel B: fused QKV GEMM, wave = 32(M) x 64(N) tile, software-pipelined
// k-loop (register double buffering). Epilogue scatters:
//   Q -> [B,H,S,hd] bf16, scaled by 1/8
//   K -> [B,H,S,hd] bf16
//   V -> [B,H,hd,S] bf16 (pre-transposed for the flash PV stage)
// NOTE: the pipeline's last-iteration prefetch over-reads 64B past a row;
// workspace region ordering guarantees this stays inside the workspace.
// ---------------------------------------------------------------------------
__global__ __launch_bounds__(128) void qkv_gemm_kernel(
    const bf16_t* __restrict__ X, const bf16_t* __restrict__ WT,
    const float* __restrict__ bias,
    bf16_t* __restrict__ Qb, bf16_t* __restrict__ Kb, bf16_t* __restrict__ VT) {
  const int tid  = threadIdx.x;
  const int wave = tid >> 5;
  const int lane = tid & 31;
  const int lm   = lane & 15;
  const int lh   = lane >> 4;

  const int w  = blockIdx.x * 4 + wave;   // 6144 waves
  const int nt = w % 48;                  // 48 N-tiles of 64
  const int mt = w / 48;                  // 128 M-tiles of 32
  const int i0 = mt * 32;
  const int j0 = nt * 64;

  v8f acc[2][4];
#pragma unroll
  for (int mi = 0; mi < 2; ++mi)
#pragma unroll
    for (int t = 0; t < 4; ++t) acc[mi][t] = v8f{};

  const bf16_t* arow0 = X + (size_t)(i0 + lm) * DIM;
  const bf16_t* arow1 = arow0 + (size_t)16 * DIM;
  const bf16_t* brow[4];
#pragma unroll
  for (int t = 0; t < 4; ++t)
    brow[t] = WT + (size_t)(j0 + t * 16 + lm) * DIM;

  v16bf a0 = load_a_frag(arow0, lh);
  v16bf a1 = load_a_frag(arow1, lh);
  v16bf b0 = load_b_frag(brow[0], lh);
  v16bf b1 = load_b_frag(brow[1], lh);
  v16bf b2 = load_b_frag(brow[2], lh);
  v16bf b3 = load_b_frag(brow[3], lh);

  for (int k = 0; k < DIM; k += 32) {
    // prefetch next-k fragments while current WMMAs run
    v16bf na0 = load_a_frag(arow0 + k + 32, lh);
    v16bf na1 = load_a_frag(arow1 + k + 32, lh);
    v16bf nb0 = load_b_frag(brow[0] + k + 32, lh);
    v16bf nb1 = load_b_frag(brow[1] + k + 32, lh);
    v16bf nb2 = load_b_frag(brow[2] + k + 32, lh);
    v16bf nb3 = load_b_frag(brow[3] + k + 32, lh);

    acc[0][0] = wmma_bf16(a0, b0, acc[0][0]);
    acc[1][0] = wmma_bf16(a1, b0, acc[1][0]);
    acc[0][1] = wmma_bf16(a0, b1, acc[0][1]);
    acc[1][1] = wmma_bf16(a1, b1, acc[1][1]);
    acc[0][2] = wmma_bf16(a0, b2, acc[0][2]);
    acc[1][2] = wmma_bf16(a1, b2, acc[1][2]);
    acc[0][3] = wmma_bf16(a0, b3, acc[0][3]);
    acc[1][3] = wmma_bf16(a1, b3, acc[1][3]);

    a0 = na0; a1 = na1; b0 = nb0; b1 = nb1; b2 = nb2; b3 = nb3;
  }

  // Epilogue: region is uniform per wave (j0 multiple of 64)
  const int region = j0 >> 10;    // 0=Q 1=K 2=V
#pragma unroll
  for (int mi = 0; mi < 2; ++mi) {
#pragma unroll
    for (int r = 0; r < 8; ++r) {
      const int i  = i0 + mi * 16 + r + 8 * lh;
      const int bb = i >> 11;
      const int s  = i & 2047;
#pragma unroll
      for (int t = 0; t < 4; ++t) {
        const int j  = j0 + t * 16 + lm;
        const float v  = acc[mi][t][r] + bias[j];
        const int jj = j & 1023;
        const int hh = jj >> 6;
        const int d  = jj & 63;
        const size_t bhh = (size_t)bb * NHEAD + hh;
        if (region == 0)
          Qb[(bhh * SEQ + s) * HD + d] = (bf16_t)(v * 0.125f);
        else if (region == 1)
          Kb[(bhh * SEQ + s) * HD + d] = (bf16_t)v;
        else
          VT[(bhh * HD + d) * SEQ + s] = (bf16_t)v;   // transposed
      }
    }
  }
}

// ---------------------------------------------------------------------------
// Async staging: copy one 64x64 bf16 tile of K (row-major [k][d]) and one of
// V^T (row-major [d][k]) into LDS with GLOBAL_LOAD_ASYNC_TO_LDS_B128.
// 128 threads x (4+4) x 16B = 16KB. Tracked by ASYNCcnt (8 per wave).
// ---------------------------------------------------------------------------
__device__ __forceinline__ void stage_tile_async(
    const bf16_t* __restrict__ Kb, const bf16_t* __restrict__ VT,
    size_t headBase, int kbase, int tid, bf16_t* sKbuf, bf16_t* sVbuf) {
  const int row = tid >> 1;          // K: k_local row ; V: d row
  const int c0  = (tid & 1) * 32;
  const bf16_t* kg = Kb + headBase + (size_t)(kbase + row) * HD + c0;
  const bf16_t* vg = VT + headBase + (size_t)row * SEQ + kbase + c0;
  bf16_t* kl = sKbuf + row * 64 + c0;
  bf16_t* vl = sVbuf + row * 64 + c0;
#pragma unroll
  for (int c = 0; c < 32; c += 8) {
    asm volatile("global_load_async_to_lds_b128 %0, %1, off"
                 :: "v"((unsigned)(uintptr_t)(kl + c)),
                    "v"((unsigned long long)(uintptr_t)(kg + c))
                 : "memory");
  }
#pragma unroll
  for (int c = 0; c < 32; c += 8) {
    asm volatile("global_load_async_to_lds_b128 %0, %1, off"
                 :: "v"((unsigned)(uintptr_t)(vl + c)),
                    "v"((unsigned long long)(uintptr_t)(vg + c))
                 : "memory");
  }
}

// ---------------------------------------------------------------------------
// Kernel C: flash attention. Block = 4 waves = 64 q rows of one (b,h).
// Double-buffered async K/V tiles; scores stay in accumulators; online
// softmax with 16-lane shuffle reductions; P relayout via wave-private LDS.
// ---------------------------------------------------------------------------
__global__ __launch_bounds__(128) void flash_attn_kernel(
    const bf16_t* __restrict__ Qb, const bf16_t* __restrict__ Kb,
    const bf16_t* __restrict__ VT, bf16_t* __restrict__ Abuf) {
  __shared__ bf16_t sK[2][64 * 64];     // [buf][k_local][d]
  __shared__ bf16_t sVt[2][64 * 64];    // [buf][d][k_local]
  __shared__ bf16_t sP[4][16 * 64];     // per-wave P staging [m][k_local]

  const int bh  = blockIdx.x;           // b*16 + h
  const int qb  = blockIdx.y * 64;
  const int tid = threadIdx.x;
  const int wave = tid >> 5;
  const int lane = tid & 31;
  const int lm = lane & 15;
  const int lh = lane >> 4;

  const int b = bh >> 4;
  const int h = bh & 15;
  const size_t headBase = (size_t)bh * SEQ * HD;
  const int q0 = qb + wave * 16;

  // Q fragments, d-chunks [0,32) and [32,64) (1/8 scale pre-folded)
  const bf16_t* qrow = Qb + headBase + (size_t)(q0 + lm) * HD;
  const v16bf qa0 = load_a_frag(qrow, lh);
  const v16bf qa1 = load_a_frag(qrow + 32, lh);

  v8f   o[4] = {v8f{}, v8f{}, v8f{}, v8f{}};
  float m[8], l[8];
#pragma unroll
  for (int r = 0; r < 8; ++r) { m[r] = -1e30f; l[r] = 0.f; }

  const int nkb = (qb >> 6) + 1;        // causal: key blocks <= q block

  // prologue: start async copy of tile 0
  stage_tile_async(Kb, VT, headBase, 0, tid, sK[0], sVt[0]);

  for (int kb = 0; kb < nkb; ++kb) {
    if (kb + 1 < nkb) {
      // issue next tile's copies, then wait for the *previous* 8 (in-order)
      stage_tile_async(Kb, VT, headBase, (kb + 1) * 64, tid,
                       sK[(kb + 1) & 1], sVt[(kb + 1) & 1]);
      asm volatile("s_wait_asynccnt 0x8" ::: "memory");
    } else {
      asm volatile("s_wait_asynccnt 0x0" ::: "memory");
    }
    __syncthreads();                    // tile kb visible to all waves

    const bf16_t* sKb = sK[kb & 1];
    const bf16_t* sVb = sVt[kb & 1];
    const int kbase = kb * 64;

    // --- S = Q @ K^T ---
    v8f s[4];
#pragma unroll
    for (int t = 0; t < 4; ++t) {
      const bf16_t* krow = &sKb[(t * 16 + lm) * 64];
      v8f c = v8f{};
      c = wmma_bf16(qa0, load_b_frag(krow, lh), c);
      c = wmma_bf16(qa1, load_b_frag(krow + 32, lh), c);
      s[t] = c;
    }

    // --- causal mask + online softmax (row = r + 8*lh, col = lm) ---
#pragma unroll
    for (int r = 0; r < 8; ++r) {
      const int qg = q0 + r + 8 * lh;
      float rowmax = -1e30f;
#pragma unroll
      for (int t = 0; t < 4; ++t) {
        const int kg = kbase + t * 16 + lm;
        float v = s[t][r];
        if (kg > qg) v = -10000.0f;     // reference MASKED_BIAS
        s[t][r] = v;
        rowmax = fmaxf(rowmax, v);
      }
#pragma unroll
      for (int off = 1; off < 16; off <<= 1)
        rowmax = fmaxf(rowmax, __shfl_xor(rowmax, off, 32));
      const float mnew  = fmaxf(m[r], rowmax);
      const float alpha = __expf(m[r] - mnew);
      float rsum = 0.f;
#pragma unroll
      for (int t = 0; t < 4; ++t) {
        float p = __expf(s[t][r] - mnew);
        s[t][r] = p;
        rsum += p;
      }
#pragma unroll
      for (int off = 1; off < 16; off <<= 1)
        rsum += __shfl_xor(rsum, off, 32);
      l[r] = alpha * l[r] + rsum;
      m[r] = mnew;
#pragma unroll
      for (int t = 0; t < 4; ++t) o[t][r] *= alpha;
#pragma unroll
      for (int t = 0; t < 4; ++t)
        sP[wave][(r + 8 * lh) * 64 + t * 16 + lm] = (bf16_t)s[t][r];
    }

    // --- O += P @ V (wave-private LDS; same-wave DS ordering) ---
    const bf16_t* prow = &sP[wave][lm * 64];
    const v16bf p0 = load_a_frag(prow, lh);
    const v16bf p1 = load_a_frag(prow + 32, lh);
#pragma unroll
    for (int t = 0; t < 4; ++t) {
      const bf16_t* vrow = &sVb[(t * 16 + lm) * 64];
      o[t] = wmma_bf16(p0, load_b_frag(vrow, lh), o[t]);
      o[t] = wmma_bf16(p1, load_b_frag(vrow + 32, lh), o[t]);
    }
    __syncthreads();    // all waves done with buf[kb&1] before it is refilled
  }

  // finalize: O /= l, merge heads -> Abuf[b][s][h*64 + d]
#pragma unroll
  for (int r = 0; r < 8; ++r) {
    const float inv = 1.0f / l[r];
    const int sg = q0 + r + 8 * lh;
#pragma unroll
    for (int t = 0; t < 4; ++t)
      Abuf[((size_t)b * SEQ + sg) * DIM + h * HD + t * 16 + lm] =
          (bf16_t)(o[t][r] * inv);
  }
}

// ---------------------------------------------------------------------------
// Kernel D: output projection, wave = 32x64 tile, pipelined like kernel B.
// ---------------------------------------------------------------------------
__global__ __launch_bounds__(128) void proj_gemm_kernel(
    const bf16_t* __restrict__ A, const bf16_t* __restrict__ WT,
    const float* __restrict__ bias, float* __restrict__ out) {
  const int tid  = threadIdx.x;
  const int wave = tid >> 5;
  const int lane = tid & 31;
  const int lm = lane & 15;
  const int lh = lane >> 4;

  const int w  = blockIdx.x * 4 + wave;   // 2048 waves
  const int nt = w % 16;
  const int mt = w / 16;                  // 128 M-tiles of 32
  const int i0 = mt * 32;
  const int j0 = nt * 64;

  v8f acc[2][4];
#pragma unroll
  for (int mi = 0; mi < 2; ++mi)
#pragma unroll
    for (int t = 0; t < 4; ++t) acc[mi][t] = v8f{};

  const bf16_t* arow0 = A + (size_t)(i0 + lm) * DIM;
  const bf16_t* arow1 = arow0 + (size_t)16 * DIM;
  const bf16_t* brow[4];
#pragma unroll
  for (int t = 0; t < 4; ++t)
    brow[t] = WT + (size_t)(j0 + t * 16 + lm) * DIM;

  v16bf a0 = load_a_frag(arow0, lh);
  v16bf a1 = load_a_frag(arow1, lh);
  v16bf b0 = load_b_frag(brow[0], lh);
  v16bf b1 = load_b_frag(brow[1], lh);
  v16bf b2 = load_b_frag(brow[2], lh);
  v16bf b3 = load_b_frag(brow[3], lh);

  for (int k = 0; k < DIM; k += 32) {
    v16bf na0 = load_a_frag(arow0 + k + 32, lh);
    v16bf na1 = load_a_frag(arow1 + k + 32, lh);
    v16bf nb0 = load_b_frag(brow[0] + k + 32, lh);
    v16bf nb1 = load_b_frag(brow[1] + k + 32, lh);
    v16bf nb2 = load_b_frag(brow[2] + k + 32, lh);
    v16bf nb3 = load_b_frag(brow[3] + k + 32, lh);

    acc[0][0] = wmma_bf16(a0, b0, acc[0][0]);
    acc[1][0] = wmma_bf16(a1, b0, acc[1][0]);
    acc[0][1] = wmma_bf16(a0, b1, acc[0][1]);
    acc[1][1] = wmma_bf16(a1, b1, acc[1][1]);
    acc[0][2] = wmma_bf16(a0, b2, acc[0][2]);
    acc[1][2] = wmma_bf16(a1, b2, acc[1][2]);
    acc[0][3] = wmma_bf16(a0, b3, acc[0][3]);
    acc[1][3] = wmma_bf16(a1, b3, acc[1][3]);

    a0 = na0; a1 = na1; b0 = nb0; b1 = nb1; b2 = nb2; b3 = nb3;
  }

#pragma unroll
  for (int mi = 0; mi < 2; ++mi) {
#pragma unroll
    for (int r = 0; r < 8; ++r) {
      const int i = i0 + mi * 16 + r + 8 * lh;
#pragma unroll
      for (int t = 0; t < 4; ++t) {
        const int j = j0 + t * 16 + lm;
        out[(size_t)i * DIM + j] = acc[mi][t][r] + bias[j];
      }
    }
  }
}

// ---------------------------------------------------------------------------
// Host launch
// ---------------------------------------------------------------------------
extern "C" void kernel_launch(void* const* d_in, const int* in_sizes, int n_in,
                              void* d_out, int out_size, void* d_ws, size_t ws_size,
                              hipStream_t stream) {
  const float* hidden   = (const float*)d_in[0];
  const float* c_attn_w = (const float*)d_in[1];
  const float* c_attn_b = (const float*)d_in[2];
  const float* c_proj_w = (const float*)d_in[3];
  const float* c_proj_b = (const float*)d_in[4];
  float* out = (float*)d_out;

  // Region order chosen so pipelined GEMM k-prefetch over-reads (<=64B past a
  // row of X / WqkvT / WpT / Abuf) land inside the next workspace region.
  char* ws = (char*)d_ws;
  size_t off = 0;
  bf16_t* X     = (bf16_t*)(ws + off); off += (size_t)ROWS * DIM * 2;      // 8 MB
  bf16_t* WqkvT = (bf16_t*)(ws + off); off += (size_t)3 * DIM * DIM * 2;   // 6 MB
  bf16_t* WpT   = (bf16_t*)(ws + off); off += (size_t)DIM * DIM * 2;       // 2 MB
  bf16_t* Abuf  = (bf16_t*)(ws + off); off += (size_t)ROWS * DIM * 2;      // 8 MB
  bf16_t* Qb    = (bf16_t*)(ws + off); off += (size_t)ROWS * DIM * 2;      // 8 MB
  bf16_t* Kb    = (bf16_t*)(ws + off); off += (size_t)ROWS * DIM * 2;      // 8 MB
  bf16_t* VT    = (bf16_t*)(ws + off); off += (size_t)ROWS * DIM * 2;      // 8 MB

  {
    int n = ROWS * DIM;
    cvt_kernel<<<(n + 255) / 256, 256, 0, stream>>>(hidden, X, n);
    int nw = DIM * 3 * DIM;
    cvt_t_kernel<<<(nw + 255) / 256, 256, 0, stream>>>(c_attn_w, WqkvT, DIM, 3 * DIM);
    int np = DIM * DIM;
    cvt_t_kernel<<<(np + 255) / 256, 256, 0, stream>>>(c_proj_w, WpT, DIM, DIM);
  }
  // 128 M-tiles(32) x 48 N-tiles(64) = 6144 waves / 4 per block
  qkv_gemm_kernel<<<1536, 128, 0, stream>>>(X, WqkvT, c_attn_b, Qb, Kb, VT);
  // (b*h) x (S/64) blocks of 4 waves
  flash_attn_kernel<<<dim3(BATCH * NHEAD, SEQ / 64), 128, 0, stream>>>(Qb, Kb, VT, Abuf);
  // 128 M-tiles(32) x 16 N-tiles(64) = 2048 waves / 4 per block
  proj_gemm_kernel<<<512, 128, 0, stream>>>(Abuf, WpT, c_proj_b, out);
}
